// GraphTransformerLayer_57131654971708
// MI455X (gfx1250) — matrix operand order repaired and checked by reading, not compiled, not run on hardware
//
#include <hip/hip_runtime.h>
#include <hip/hip_bf16.h>

// ---------------------------------------------------------------------------
// GraphTransformerLayer on gfx1250 (MI455X), sparse-softmax reformulation.
// GEMMs use v_wmma_f32_16x16x32_f16 (f16 A/B, f32 accum).
// ---------------------------------------------------------------------------

typedef _Float16 half_t;
typedef __attribute__((ext_vector_type(16))) _Float16 v16h;
typedef __attribute__((ext_vector_type(8)))  _Float16 v8h;
typedef __attribute__((ext_vector_type(8)))  float    v8f;

#define HID 128

// ---------------- converts ----------------
__global__ void f32_to_f16_kernel(const float* __restrict__ in, half_t* __restrict__ out, int n) {
  int i = blockIdx.x * 256 + threadIdx.x;
  if (i < n) out[i] = (half_t)in[i];
}

// W[K,Ncols] (row-major f32) -> WT[Ncols,K] f16 (so B-fragments are contiguous)
__global__ void transpose_f16_kernel(const float* __restrict__ W, half_t* __restrict__ WT,
                                     int K, int Ncols) {
  int i = blockIdx.x * 256 + threadIdx.x;
  if (i < K * Ncols) {
    int k = i / Ncols, n = i % Ncols;
    WT[(size_t)n * K + k] = (half_t)W[i];
  }
}

// ---------------- WMMA GEMM: C[M,Ncols] = A[M,K] @ BT^T + bias ----------------
// mode 0: f16 out; 1: f16 out + relu; 2: f32 out; 3: f32 out + residual add
__global__ void gemm_wmma_kernel(const half_t* __restrict__ A, const half_t* __restrict__ BT,
                                 const float* __restrict__ bias, void* __restrict__ Cout,
                                 const float* __restrict__ residual,
                                 int M, int K, int Ncols, int mode) {
  const int lane = threadIdx.x & 31;
  const int wave = threadIdx.x >> 5;
  const int row0 = blockIdx.x * 16;
  const int col0 = blockIdx.y * 128 + wave * 16;
  if (col0 >= Ncols || row0 >= M) return;

  const int rowA  = row0 + (lane & 15);
  const int kbase = (lane >> 4) * 8;    // A elems 0..7 -> K kbase..kbase+7; 8..15 -> +16
  const int colB  = col0 + (lane & 15);
  const int kboff = (lane >> 4) * 8;    // B elems 0..7 -> K kboff..kboff+7; 8..15 -> +16

  const half_t* __restrict__ Arow = A  + (size_t)rowA * K;
  const half_t* __restrict__ Bcol = BT + (size_t)colB * K;

  v8f acc = {};
  for (int k0 = 0; k0 < K; k0 += 32) {
    __builtin_prefetch(Arow + k0 + 64, 0, 1);   // global_prefetch_b8
    v8h alo = *(const v8h*)(Arow + k0 + kbase);
    v8h ahi = *(const v8h*)(Arow + k0 + 16 + kbase);
    v16h a;
#pragma unroll
    for (int i = 0; i < 8; ++i) { a[i] = alo[i]; a[i + 8] = ahi[i]; }
    v8h b0 = *(const v8h*)(Bcol + k0 + kboff);
    v8h b1 = *(const v8h*)(Bcol + k0 + 16 + kboff);
    v16h b;
#pragma unroll
    for (int i = 0; i < 8; ++i) { b[i] = b0[i]; b[i + 8] = b1[i]; }
    acc = __builtin_amdgcn_wmma_f32_16x16x32_f16(false, a, false, b, (short)0, acc,
                                                 false, false);
  }

  const int col   = colB;
  const float bv  = bias ? bias[col] : 0.0f;
  const int rbase = row0 + (lane >> 4) * 8;

  if (mode <= 1) {
    half_t* C = (half_t*)Cout;
#pragma unroll
    for (int v = 0; v < 8; ++v) {
      float val = acc[v] + bv;
      if (mode == 1) val = val > 0.0f ? val : 0.0f;
      C[(size_t)(rbase + v) * Ncols + col] = (half_t)val;
    }
  } else {
    float* C = (float*)Cout;
#pragma unroll
    for (int v = 0; v < 8; ++v) {
      float val = acc[v] + bv;
      if (mode == 3) val += residual[(size_t)(rbase + v) * Ncols + col];
      C[(size_t)(rbase + v) * Ncols + col] = val;
    }
  }
}

// ---------------- edge bias: eb[e] = edge_attr[e,:] . We + be ----------------
__global__ void ebias_kernel(const float* __restrict__ EA, const float* __restrict__ We,
                             const float* __restrict__ be, float* __restrict__ eb, int E) {
  int e = blockIdx.x * 256 + threadIdx.x;
  if (e >= E) return;
  float s = be[0];
#pragma unroll
  for (int j = 0; j < 16; ++j) s += EA[(size_t)e * 16 + j] * We[j];
  eb[e] = s;
}

// ---------------- per-edge score (one wave32 per edge) ----------------
__global__ void score_kernel(const half_t* __restrict__ Qh, const half_t* __restrict__ Kh,
                             const long long* __restrict__ ei, const float* __restrict__ eb,
                             float* __restrict__ s, float* __restrict__ m, int E) {
  int gid  = blockIdx.x * 256 + threadIdx.x;
  int lane = gid & 31;
  int e    = gid >> 5;
  if (e >= E) return;
  long long src = ei[e];
  long long dst = ei[(size_t)E + e];
  const half_t* q = Qh + (size_t)src * HID + lane * 4;
  const half_t* k = Kh + (size_t)dst * HID + lane * 4;
  float p = 0.0f;
#pragma unroll
  for (int j = 0; j < 4; ++j) p += (float)q[j] * (float)k[j];
#pragma unroll
  for (int off = 16; off > 0; off >>= 1) p += __shfl_xor(p, off, 32);
  if (lane == 0) {
    float sc = p + eb[e];
    sc = (sc >= 0.0f) ? sc : 0.01f * sc;   // leaky relu
    s[e] = sc;
    // row max has an implicit 0 floor (non-edges); positive floats order as ints
    if (sc > 0.0f) atomicMax((int*)&m[src], __float_as_int(sc));
  }
}

// ---------------- counting sort of edges by src ----------------
__global__ void deg_kernel(const long long* __restrict__ ei, int* __restrict__ deg, int E) {
  int e = blockIdx.x * 256 + threadIdx.x;
  if (e < E) atomicAdd(&deg[ei[e]], 1);
}

__global__ void prefix_kernel(const int* __restrict__ deg, int* __restrict__ off,
                              int* __restrict__ cursor, int N, int E) {
  __shared__ int part[1024];
  int t = threadIdx.x;               // 1024 threads
  int per = N / 1024;
  int base = t * per;
  int s = 0;
  for (int i = 0; i < per; ++i) s += deg[base + i];
  part[t] = s;
  __syncthreads();
  for (int d = 1; d < 1024; d <<= 1) {
    int v = (t >= d) ? part[t - d] : 0;
    __syncthreads();
    part[t] += v;
    __syncthreads();
  }
  int run = (t == 0) ? 0 : part[t - 1];
  for (int i = 0; i < per; ++i) {
    off[base + i] = run;
    cursor[base + i] = run;
    run += deg[base + i];
  }
  if (t == 1023) off[N] = E;
}

__global__ void scatter_kernel(const long long* __restrict__ ei, int* __restrict__ cursor,
                               int* __restrict__ sorted, int E) {
  int e = blockIdx.x * 256 + threadIdx.x;
  if (e >= E) return;
  int p = atomicAdd(&cursor[ei[e]], 1);
  sorted[p] = e;
}

// ---------------- column sums (and optional sum of squares) ----------------
__global__ void colsum_kernel(const float* __restrict__ X, float* __restrict__ sums,
                              float* __restrict__ sumsq, int N) {
  int c = threadIdx.x;               // 128 threads
  int r0 = blockIdx.x * 128;
  float s = 0.0f, q = 0.0f;
  for (int r = 0; r < 128; ++r) {
    float v = X[(size_t)(r0 + r) * HID + c];
    s += v;
    if (sumsq) q += v * v;
  }
  atomicAdd(&sums[c], s);
  if (sumsq) atomicAdd(&sumsq[c], q);
}

// ---------------- sparse softmax aggregation: r1 = x + attn @ V ----------------
__global__ void aggregate_kernel(const float* __restrict__ x, const float* __restrict__ V,
                                 const float* __restrict__ s, const float* __restrict__ m,
                                 const float* __restrict__ sumV,
                                 const long long* __restrict__ ei,
                                 const int* __restrict__ off, const int* __restrict__ sorted,
                                 float* __restrict__ r1, int N, int E) {
  __shared__ float wS[128];
  __shared__ int   dS[128];
  __shared__ float zS[128];
  int i = blockIdx.x;
  int c = threadIdx.x;               // 128 threads = channels
  float mi = m[i];
  float e0 = __expf(-mi);            // weight of each of the (N-deg) zero entries
  int b = off[i], eend = off[i + 1];
  float acc = 0.0f, zp = 0.0f;
  for (int base = b; base < eend; base += 128) {
    int idx = base + c;
    float w = 0.0f;
    int d = 0;
    if (idx < eend) {
      int e = sorted[idx];
      d = (int)ei[(size_t)E + e];
      w = __expf(s[e] - mi) - e0;    // delta vs the zero-entry weight
      zp += w;
    }
    wS[c] = w;
    dS[c] = d;
    __syncthreads();
    int cnt = min(128, eend - base);
    for (int k = 0; k < cnt; ++k) acc += wS[k] * V[(size_t)dS[k] * HID + c];
    __syncthreads();
  }
  zS[c] = zp;
  __syncthreads();
  for (int d2 = 64; d2 > 0; d2 >>= 1) {
    if (c < d2) zS[c] += zS[c + d2];
    __syncthreads();
  }
  float Z = zS[0] + (float)N * e0;
  float numer = acc + e0 * sumV[c];
  r1[(size_t)i * HID + c] = x[(size_t)i * HID + c] + numer / Z;
}

// ---------------- batchnorm finalize + apply ----------------
__global__ void bnfin_kernel(const float* __restrict__ sums, const float* __restrict__ sumsq,
                             const float* __restrict__ g, const float* __restrict__ bta,
                             float* __restrict__ scale, float* __restrict__ shift, int N) {
  int c = threadIdx.x;               // 128 threads
  float mu  = sums[c] / (float)N;
  float var = sumsq[c] / (float)N - mu * mu;
  float sc  = g[c] * rsqrtf(var + 1e-5f);
  scale[c] = sc;
  shift[c] = bta[c] - mu * sc;
}

__global__ void bnapply_kernel(const float* __restrict__ in, const float* __restrict__ scale,
                               const float* __restrict__ shift, float* __restrict__ out,
                               half_t* __restrict__ outh, int total) {
  int i = blockIdx.x * 256 + threadIdx.x;
  if (i >= total) return;
  int c = i & (HID - 1);
  float v = in[i] * scale[c] + shift[c];
  out[i] = v;
  if (outh) outh[i] = (half_t)v;
}

// ---------------------------------------------------------------------------
extern "C" void kernel_launch(void* const* d_in, const int* in_sizes, int n_in,
                              void* d_out, int out_size, void* d_ws, size_t ws_size,
                              hipStream_t stream) {
  const float*     x    = (const float*)d_in[0];
  const long long* ei   = (const long long*)d_in[1];
  const float*     ea   = (const float*)d_in[2];
  const float*     Wq   = (const float*)d_in[3];
  const float*     bq   = (const float*)d_in[4];
  const float*     Wk   = (const float*)d_in[5];
  const float*     bk   = (const float*)d_in[6];
  const float*     Wv   = (const float*)d_in[7];
  const float*     bv   = (const float*)d_in[8];
  const float*     We   = (const float*)d_in[9];
  const float*     be   = (const float*)d_in[10];
  const float*     bn1g = (const float*)d_in[11];
  const float*     bn1b = (const float*)d_in[12];
  const float*     W1   = (const float*)d_in[13];
  const float*     b1   = (const float*)d_in[14];
  const float*     W2   = (const float*)d_in[15];
  const float*     b2   = (const float*)d_in[16];
  const float*     bn2g = (const float*)d_in[17];
  const float*     bn2b = (const float*)d_in[18];

  const int N = in_sizes[0] / HID;     // 16384
  const int E = in_sizes[1] / 2;       // 524288
  const int H2 = 2 * HID;              // 256

  // ---- workspace carve-out (256B aligned) ----
  char* wsb = (char*)d_ws;
  size_t pos = 0;
  auto alloc = [&](size_t bytes) -> void* {
    pos = (pos + 255) & ~(size_t)255;
    void* p = wsb + pos;
    pos += bytes;
    return p;
  };
  half_t* xh   = (half_t*)alloc((size_t)N * HID * 2);
  half_t* WqT  = (half_t*)alloc((size_t)HID * HID * 2);
  half_t* WkT  = (half_t*)alloc((size_t)HID * HID * 2);
  half_t* WvT  = (half_t*)alloc((size_t)HID * HID * 2);
  half_t* W1T  = (half_t*)alloc((size_t)HID * H2 * 2);
  half_t* W2T  = (half_t*)alloc((size_t)H2 * HID * 2);
  half_t* Qh   = (half_t*)alloc((size_t)N * HID * 2);
  half_t* Kh   = (half_t*)alloc((size_t)N * HID * 2);
  float*  Vf   = (float*)alloc((size_t)N * HID * 4);
  float*  eb   = (float*)alloc((size_t)E * 4);
  float*  sarr = (float*)alloc((size_t)E * 4);
  // --- zeroed region: m, deg, stats (contiguous, sizes are 256B multiples) ---
  float*  marr = (float*)alloc((size_t)N * 4);
  int*    deg  = (int*)alloc((size_t)N * 4);
  float*  stats = (float*)alloc((size_t)16 * HID * 4);
  size_t  zero_bytes = (size_t)N * 4 + (size_t)N * 4 + (size_t)16 * HID * 4;
  // ---------------------------------------------------------------------------
  int*    offA   = (int*)alloc((size_t)(N + 1) * 4);
  int*    cursor = (int*)alloc((size_t)N * 4);
  int*    sorted = (int*)alloc((size_t)E * 4);
  float*  r1     = (float*)alloc((size_t)N * HID * 4);   // x+attnV, later h+ffn
  float*  hbuf   = (float*)alloc((size_t)N * HID * 4);
  half_t* hh     = (half_t*)alloc((size_t)N * HID * 2);
  half_t* mid    = (half_t*)alloc((size_t)N * H2 * 2);

  float* sum1 = stats, *sq1 = stats + 128, *scale1 = stats + 256, *shift1 = stats + 384;
  float* sum2 = stats + 512, *sq2 = stats + 640, *scale2 = stats + 768, *shift2 = stats + 896;
  float* sumV = stats + 1024;

  hipMemsetAsync(marr, 0, zero_bytes, stream);

  // converts + weight transposes
  f32_to_f16_kernel<<<(N * HID + 255) / 256, 256, 0, stream>>>(x, xh, N * HID);
  transpose_f16_kernel<<<(HID * HID + 255) / 256, 256, 0, stream>>>(Wq, WqT, HID, HID);
  transpose_f16_kernel<<<(HID * HID + 255) / 256, 256, 0, stream>>>(Wk, WkT, HID, HID);
  transpose_f16_kernel<<<(HID * HID + 255) / 256, 256, 0, stream>>>(Wv, WvT, HID, HID);
  transpose_f16_kernel<<<(HID * H2 + 255) / 256, 256, 0, stream>>>(W1, W1T, HID, H2);
  transpose_f16_kernel<<<(H2 * HID + 255) / 256, 256, 0, stream>>>(W2, W2T, H2, HID);

  // QKV GEMMs (WMMA)
  dim3 g1(N / 16, 1);
  gemm_wmma_kernel<<<g1, 256, 0, stream>>>(xh, WqT, bq, (void*)Qh, nullptr, N, HID, HID, 0);
  gemm_wmma_kernel<<<g1, 256, 0, stream>>>(xh, WkT, bk, (void*)Kh, nullptr, N, HID, HID, 0);
  gemm_wmma_kernel<<<g1, 256, 0, stream>>>(xh, WvT, bv, (void*)Vf, nullptr, N, HID, HID, 2);

  // edge pipeline
  ebias_kernel<<<(E + 255) / 256, 256, 0, stream>>>(ea, We, be, eb, E);
  score_kernel<<<(E * 32 + 255) / 256, 256, 0, stream>>>(Qh, Kh, ei, eb, sarr, marr, E);
  deg_kernel<<<(E + 255) / 256, 256, 0, stream>>>(ei, deg, E);
  prefix_kernel<<<1, 1024, 0, stream>>>(deg, offA, cursor, N, E);
  scatter_kernel<<<(E + 255) / 256, 256, 0, stream>>>(ei, cursor, sorted, E);
  colsum_kernel<<<N / 128, 128, 0, stream>>>(Vf, sumV, nullptr, N);
  aggregate_kernel<<<N, 128, 0, stream>>>(x, Vf, sarr, marr, sumV, ei, offA, sorted, r1, N, E);

  // BN1
  colsum_kernel<<<N / 128, 128, 0, stream>>>(r1, sum1, sq1, N);
  bnfin_kernel<<<1, 128, 0, stream>>>(sum1, sq1, bn1g, bn1b, scale1, shift1, N);
  bnapply_kernel<<<(N * HID + 255) / 256, 256, 0, stream>>>(r1, scale1, shift1, hbuf, hh, N * HID);

  // FFN (WMMA): mid = relu(h@W1+b1); t2 = mid@W2 + b2 + h  (t2 reuses r1)
  dim3 g2(N / 16, 2);
  gemm_wmma_kernel<<<g2, 256, 0, stream>>>(hh, W1T, b1, (void*)mid, nullptr, N, HID, H2, 1);
  gemm_wmma_kernel<<<g1, 256, 0, stream>>>(mid, W2T, b2, (void*)r1, hbuf, N, H2, HID, 3);

  // BN2 -> d_out
  colsum_kernel<<<N / 128, 128, 0, stream>>>(r1, sum2, sq2, N);
  bnfin_kernel<<<1, 128, 0, stream>>>(sum2, sq2, bn2g, bn2b, scale2, shift2, N);
  bnapply_kernel<<<(N * HID + 255) / 256, 256, 0, stream>>>(r1, scale2, shift2, (float*)d_out,
                                                            nullptr, N * HID);
  (void)n_in; (void)out_size; (void)ws_size;
}
